// Llama4DecoderLayer_33913061769722
// MI455X (gfx1250) — matrix operand order repaired and checked by reading, not compiled, not run on hardware
//
#include <hip/hip_runtime.h>
#include <hip/hip_bf16.h>
#include <stdint.h>

// ---------------------------------------------------------------------------
// Types for CDNA5 WMMA (wave32): bf16 16x16x32, fp32 accumulate.
// ---------------------------------------------------------------------------
typedef __attribute__((ext_vector_type(16))) __bf16       v16bf;
typedef __attribute__((ext_vector_type(8)))  float        v8f;
typedef __attribute__((ext_vector_type(8)))  unsigned int v8u;
typedef __attribute__((ext_vector_type(4)))  int          v4i;

static __device__ __forceinline__ v8f wmma_bf16(v16bf a, v16bf b, v8f c) {
  return __builtin_amdgcn_wmma_f32_16x16x32_bf16(false, a, false, b, (short)0, c,
                                                 false, false);
}

// Pack two fp32 -> packed bf16x2 (lo -> bits[15:0]) : lowers to v_cvt_pk_bf16_f32
static __device__ __forceinline__ unsigned pack_bf16x2(float lo, float hi) {
  union { __bf16 h[2]; unsigned u; } un;
  un.h[0] = (__bf16)lo;
  un.h[1] = (__bf16)hi;
  return un.u;
}

// ---------------------------------------------------------------------------
// Async global->LDS copy (16B per lane) with sync fallback.
// ---------------------------------------------------------------------------
#if defined(__has_builtin)
#if __has_builtin(__builtin_amdgcn_global_load_async_to_lds_b128)
#define HAVE_ASYNC_LDS 1
#endif
#endif

typedef __attribute__((address_space(1))) v4i gv4i;
typedef __attribute__((address_space(3))) v4i lv4i;

static __device__ __forceinline__ void cp16_lds(void* l, const void* g) {
#ifdef HAVE_ASYNC_LDS
  __builtin_amdgcn_global_load_async_to_lds_b128(
      (gv4i*)const_cast<void*>(g), (lv4i*)l, 0, 0);
#else
  *(uint4*)l = *(const uint4*)g;
#endif
}

static __device__ __forceinline__ void wait_lds_copies() {
#ifdef HAVE_ASYNC_LDS
  asm volatile("s_wait_asynccnt 0" ::: "memory");
#endif
}

constexpr int T_   = 2048;
constexpr int H_   = 2048;
constexpr int NH_  = 16;
constexpr int NKV_ = 4;
constexpr int HD_  = 128;
constexpr int E_   = 8;
constexpr int I_   = 4096;
constexpr int QS_  = NH_ * HD_;          // 2048
constexpr int KVS_ = NKV_ * HD_;         // 512
constexpr int QKV_ = QS_ + 2 * KVS_;     // 3072
constexpr float EPS_   = 1e-5f;
constexpr float THETA_ = 500000.0f;

// ---------------------------------------------------------------------------
// RMSNorm over a row slice, output bf16.
// ---------------------------------------------------------------------------
__global__ __launch_bounds__(256)
void rmsnorm_bf16(const float* __restrict__ x, int xld, int xoff,
                  const float* __restrict__ w, __bf16* __restrict__ out,
                  int old_, int width) {
  int t = blockIdx.x, tid = threadIdx.x;
  const float* row = x + (size_t)t * xld + xoff;
  float ss = 0.f;
  for (int i = tid; i < width; i += 256) { float v = row[i]; ss += v * v; }
  __shared__ float red[256];
  red[tid] = ss; __syncthreads();
  for (int s = 128; s > 0; s >>= 1) { if (tid < s) red[tid] += red[tid + s]; __syncthreads(); }
  __shared__ float rinv;
  if (tid == 0) rinv = rsqrtf(red[0] / (float)width + EPS_);
  __syncthreads();
  float ri = rinv;
  for (int i = tid; i < width; i += 256)
    out[(size_t)t * old_ + i] = (__bf16)(row[i] * ri * w[i]);
}

// ---------------------------------------------------------------------------
// RoPE in-place on q and k slices of the fp32 qkv buffer.
// grid: (T, NH+NKV), block 64 (= half of HD)
// ---------------------------------------------------------------------------
__global__ __launch_bounds__(64)
void rope_kernel(const int* __restrict__ pos, float* __restrict__ qkv) {
  int t = blockIdx.x, hb = blockIdx.y, i = threadIdx.x;
  int col = (hb < NH_) ? hb * HD_ : QS_ + (hb - NH_) * HD_;
  float* base = qkv + (size_t)t * QKV_ + col;
  float p = (float)pos[t];
  float inv = 1.0f / __powf(THETA_, (float)i / 64.0f);
  float fr = p * inv;
  float c = __cosf(fr), s = __sinf(fr);
  float x1 = base[i], x2 = base[i + 64];
  base[i]      = x1 * c - x2 * s;
  base[i + 64] = x2 * c + x1 * s;
}

// fp32 v slice of qkv -> bf16 buffer
__global__ __launch_bounds__(256)
void conv_v(const float* __restrict__ qkv, __bf16* __restrict__ vbf) {
  int i = blockIdx.x * 256 + threadIdx.x;
  if (i < T_ * KVS_) {
    int t = i >> 9, c = i & 511;
    vbf[i] = (__bf16)qkv[(size_t)t * QKV_ + QS_ + KVS_ + c];
  }
}

// ---------------------------------------------------------------------------
// Generic bf16-WMMA GEMM: C[M x N](fp32) = A[M x K](bf16) * B[K x N](fp32->bf16)
// Block tile 128x128, K-step 32. 8 waves in a 4x2 layout; each wave owns a
// 32x64 sub-tile = 8 WMMA accumulators (2 A frags x 4 B frags per k-step).
// A tiles staged with async global->LDS (b128/lane) using loop-invariant,
// pre-gathered row pointers; B tiles converted fp32->bf16 with packed dword
// DS stores; next-k B rows prefetched (global_prefetch_b8).
// Optional: A-row gather, device row-base, device M, C-row scatter, per-C-row
// scale, elementwise addend (same layout as C), accumulate into C.
// ---------------------------------------------------------------------------
__global__ __launch_bounds__(256)
void gemm_wmma(const __bf16* __restrict__ A, const float* __restrict__ B,
               float* __restrict__ C, int M, int N, int K,
               int lda, int ldb, int ldc,
               const int* a_rows, const int* base_ptr, const int* m_ptr,
               const int* c_rows, const float* c_scale,
               const float* add, int accFlag) {
  int tid  = threadIdx.x;
  int wave = tid >> 5, lane = tid & 31;
  int hi = lane >> 4, ln = lane & 15;
  int wr = wave & 3, wc = wave >> 2;

  int mEff = m_ptr ? *m_ptr : M;
  int base = base_ptr ? *base_ptr : 0;
  int rowTile = blockIdx.y * 128;
  if (rowTile >= mEff) return;               // uniform exit
  int colTile = blockIdx.x * 128;

  __shared__ __bf16 As[128][40];             // [row][k], padded (80B rows, 16B-aligned)
  __shared__ __bf16 Bs[128][40];             // [col][k] (k-pairs contiguous)

  // Loop-invariant A staging state: this thread always stages rows r0 and
  // 64+r0 at column chunk c0. Gather lookups + zero-fill happen once.
  int r0 = tid >> 2;                         // 0..63
  int c0 = (tid & 3) * 8;                    // 0,8,16,24
  int rrA = rowTile + r0;
  int rrB = rowTile + 64 + r0;
  int ar0 = (rrA < mEff) ? (a_rows ? a_rows[rrA] : (base + rrA)) : -1;
  int ar1 = (rrB < mEff) ? (a_rows ? a_rows[rrB] : (base + rrB)) : -1;
  const __bf16* srcA0 = A + (size_t)(ar0 < 0 ? 0 : ar0) * lda + c0;
  const __bf16* srcA1 = A + (size_t)(ar1 < 0 ? 0 : ar1) * lda + c0;
  if (ar0 < 0) { uint4 z = {0, 0, 0, 0}; *(uint4*)&As[r0][c0]      = z; }
  if (ar1 < 0) { uint4 z = {0, 0, 0, 0}; *(uint4*)&As[64 + r0][c0] = z; }
  __syncthreads();

  v8f acc[8] = {};

  for (int k0 = 0; k0 < K; k0 += 32) {
    // stage A tile: 128x32 bf16, 16B per lane (async, pointers hoisted)
    if (ar0 >= 0) cp16_lds(&As[r0][c0],      srcA0 + k0);
    if (ar1 >= 0) cp16_lds(&As[64 + r0][c0], srcA1 + k0);
    // stage B tile: 32(k) x 128(col) fp32 -> bf16 packed pairs into Bs[col][k]
#pragma unroll
    for (int it = 0; it < 2; ++it) {
      int idx = it * 256 + tid;
      int k = (idx & 15) * 2;
      int c = (idx >> 4) * 4;
      const float4 v0 = *(const float4*)&B[(size_t)(k0 + k)     * ldb + colTile + c];
      const float4 v1 = *(const float4*)&B[(size_t)(k0 + k + 1) * ldb + colTile + c];
      *(unsigned*)&Bs[c + 0][k] = pack_bf16x2(v0.x, v1.x);
      *(unsigned*)&Bs[c + 1][k] = pack_bf16x2(v0.y, v1.y);
      *(unsigned*)&Bs[c + 2][k] = pack_bf16x2(v0.z, v1.z);
      *(unsigned*)&Bs[c + 3][k] = pack_bf16x2(v0.w, v1.w);
    }
    // prefetch next k-step's B rows (global_prefetch_b8)
    if (k0 + 32 < K) {
      int k = (tid & 15) * 2;
      int c = (tid >> 4) * 4;
      __builtin_prefetch(&B[(size_t)(k0 + 32 + k) * ldb + colTile + c], 0, 3);
    }
    wait_lds_copies();
    __syncthreads();

    // A fragments (rows wr*32 .. wr*32+31): kBase = hi*16
    const unsigned int* arow0 = (const unsigned int*)&As[wr * 32 + ln][0];
    const unsigned int* arow1 = (const unsigned int*)&As[wr * 32 + 16 + ln][0];
    v8u au0, au1;
#pragma unroll
    for (int p = 0; p < 8; ++p) {
      int kk = hi * 16 + p * 2;
      au0[p] = arow0[kk >> 1];
      au1[p] = arow1[kk >> 1];
    }
    v16bf af0 = __builtin_bit_cast(v16bf, au0);
    v16bf af1 = __builtin_bit_cast(v16bf, au1);

#pragma unroll
    for (int ct = 0; ct < 4; ++ct) {
      const unsigned int* brow = (const unsigned int*)&Bs[wc * 64 + ct * 16 + ln][0];
      v8u bu;
#pragma unroll
      for (int p = 0; p < 8; ++p) bu[p] = brow[hi * 8 + p];
      v16bf bf = __builtin_bit_cast(v16bf, bu);
      acc[ct]     = wmma_bf16(af0, bf, acc[ct]);
      acc[4 + ct] = wmma_bf16(af1, bf, acc[4 + ct]);
    }
    __syncthreads();
  }

  // epilogue: C tile layout -> row = e + hi*8 within 16-tile, col = ln
#pragma unroll
  for (int rt = 0; rt < 2; ++rt) {
#pragma unroll
    for (int e = 0; e < 8; ++e) {
      int r = rowTile + wr * 32 + rt * 16 + hi * 8 + e;
      if (r < mEff) {
        int crow = c_rows ? c_rows[r] : (base + r);
        float sc = c_scale ? c_scale[crow] : 1.0f;
#pragma unroll
        for (int ct = 0; ct < 4; ++ct) {
          int col = colTile + wc * 64 + ct * 16 + ln;
          size_t off = (size_t)crow * ldc + col;
          float v = acc[rt * 4 + ct][e] * sc;
          if (add)     v += add[off];
          if (accFlag) v += C[off];
          C[off] = v;
        }
      }
    }
  }
}

// ---------------------------------------------------------------------------
// Flash attention (causal, GQA 16q/4kv heads, HD=128), bf16 WMMA, fp32 softmax.
// grid (T/128, NH), 256 threads. Each wave owns 16 query rows.
// ---------------------------------------------------------------------------
__global__ __launch_bounds__(256)
void flash_attn(const __bf16* __restrict__ Qb, const __bf16* __restrict__ Kb,
                const __bf16* __restrict__ Vb, __bf16* __restrict__ Ob) {
  const float scale = 0.08838834764831845f;  // 1/sqrt(128)
  int h = blockIdx.y, kvh = h >> 2;
  int qBase = blockIdx.x * 128;
  int tid = threadIdx.x, wave = tid >> 5, lane = tid & 31;
  int hi = lane >> 4, ln = lane & 15;

  __shared__ __bf16 Qs[128][136];   // [q][d]
  __shared__ __bf16 Ks[32][136];    // [key][d]  (acts as B[col=key][k=d])
  __shared__ __bf16 Vst[128][40];   // [d][key]  (acts as B[col=d][k=key])
  __shared__ __bf16 Ps[8][16][40];  // per-wave P tile [q][key]

  // load Q block once (async)
#pragma unroll
  for (int it = 0; it < 8; ++it) {
    int f = it * 2048 + tid * 8;
    int q = f >> 7, d = f & 127;
    cp16_lds(&Qs[q][d], &Qb[(size_t)(qBase + q) * QS_ + h * HD_ + d]);
  }

  v8f o[8] = {};
  float m_[8], l_[8];
#pragma unroll
  for (int e = 0; e < 8; ++e) { m_[e] = -__builtin_inff(); l_[e] = 0.f; }

  int nkb = (qBase + 128) >> 5;
  for (int kb = 0; kb < nkb; ++kb) {
    __syncthreads();
#pragma unroll
    for (int it = 0; it < 2; ++it) {           // K tile: 32x128 (async)
      int f = it * 2048 + tid * 8;
      int key = f >> 7, d = f & 127;
      cp16_lds(&Ks[key][d],
               &Kb[(size_t)((kb << 5) + key) * KVS_ + kvh * HD_ + d]);
    }
    for (int i = tid; i < 4096; i += 256) {    // V tile transposed (scalar)
      int key = i >> 7, d = i & 127;
      Vst[d][key] = Vb[(size_t)((kb << 5) + key) * KVS_ + kvh * HD_ + d];
    }
    wait_lds_copies();
    __syncthreads();

    // S = Q * K^T  (two 16-key tiles, 4 k-steps over HD=128)
    v8f s0 = {}, s1 = {};
#pragma unroll
    for (int ks = 0; ks < 4; ++ks) {
      const unsigned int* qrow = (const unsigned int*)&Qs[wave * 16 + ln][0];
      v8u au;
#pragma unroll
      for (int p = 0; p < 8; ++p) {
        int kk = ks * 32 + hi * 16 + p * 2;
        au[p] = qrow[kk >> 1];
      }
      v16bf af = __builtin_bit_cast(v16bf, au);

      const unsigned int* b0 = (const unsigned int*)&Ks[ln][0];
      const unsigned int* b1 = (const unsigned int*)&Ks[16 + ln][0];
      v8u bu0, bu1;
#pragma unroll
      for (int p = 0; p < 8; ++p) {
        bu0[p] = b0[ks * 16 + hi * 8 + p];
        bu1[p] = b1[ks * 16 + hi * 8 + p];
      }
      s0 = wmma_bf16(af, __builtin_bit_cast(v16bf, bu0), s0);
      s1 = wmma_bf16(af, __builtin_bit_cast(v16bf, bu1), s1);
    }

    // online softmax update (rows spread across 16 lanes of each half-wave)
#pragma unroll
    for (int e = 0; e < 8; ++e) {
      int q = qBase + wave * 16 + hi * 8 + e;
      int key0 = (kb << 5) + ln, key1 = key0 + 16;
      float v0 = (key0 <= q) ? s0[e] * scale : -__builtin_inff();
      float v1 = (key1 <= q) ? s1[e] * scale : -__builtin_inff();
      float t = fmaxf(v0, v1);
      t = fmaxf(t, __shfl_xor(t, 1));
      t = fmaxf(t, __shfl_xor(t, 2));
      t = fmaxf(t, __shfl_xor(t, 4));
      t = fmaxf(t, __shfl_xor(t, 8));
      float mn = fmaxf(m_[e], t);
      float mc = (mn == -__builtin_inff()) ? 0.f : mn;
      float p0 = __expf(v0 - mc), p1 = __expf(v1 - mc);
      float f = __expf(m_[e] - mc);
      float rs = p0 + p1;
      rs += __shfl_xor(rs, 1);
      rs += __shfl_xor(rs, 2);
      rs += __shfl_xor(rs, 4);
      rs += __shfl_xor(rs, 8);
      l_[e] = l_[e] * f + rs;
      m_[e] = mn;
#pragma unroll
      for (int ct = 0; ct < 8; ++ct) o[ct][e] *= f;
      Ps[wave][hi * 8 + e][ln]      = (__bf16)p0;
      Ps[wave][hi * 8 + e][16 + ln] = (__bf16)p1;
    }

    // O += P * V   (A = P 16x32, B = V 32x16 per dim-tile; same-wave LDS RAW ok)
    const unsigned int* prow = (const unsigned int*)&Ps[wave][ln][0];
    v8u pu;
#pragma unroll
    for (int p = 0; p < 8; ++p) {
      int kk = hi * 16 + p * 2;
      pu[p] = prow[kk >> 1];
    }
    v16bf pf = __builtin_bit_cast(v16bf, pu);
#pragma unroll
    for (int ct = 0; ct < 8; ++ct) {
      const unsigned int* vrow = (const unsigned int*)&Vst[ct * 16 + ln][0];
      v8u bu;
#pragma unroll
      for (int p = 0; p < 8; ++p) bu[p] = vrow[hi * 8 + p];
      o[ct] = wmma_bf16(pf, __builtin_bit_cast(v16bf, bu), o[ct]);
    }
  }

  // normalize and emit bf16 for the o-proj GEMM
#pragma unroll
  for (int e = 0; e < 8; ++e) {
    float inv = (l_[e] > 0.f) ? 1.f / l_[e] : 0.f;
    int q = qBase + wave * 16 + hi * 8 + e;
#pragma unroll
    for (int ct = 0; ct < 8; ++ct)
      Ob[(size_t)q * QS_ + h * HD_ + ct * 16 + ln] = (__bf16)(o[ct][e] * inv);
  }
}

// ---------------------------------------------------------------------------
// Router: logits = xn2 @ router_w (H x 8), top-1 + sigmoid gate.
// ---------------------------------------------------------------------------
__global__ __launch_bounds__(256)
void router_kernel(const __bf16* __restrict__ xn2, const float* __restrict__ rw,
                   int* __restrict__ top_idx, float* __restrict__ top_gate) {
  int t = blockIdx.x, tid = threadIdx.x;
  float acc[E_] = {};
  for (int i = tid; i < H_; i += 256) {
    float x = (float)xn2[(size_t)t * H_ + i];
#pragma unroll
    for (int e = 0; e < E_; ++e) acc[e] += x * rw[(size_t)i * E_ + e];
  }
  __shared__ float red[256];
  __shared__ float logits[E_];
  for (int e = 0; e < E_; ++e) {
    red[tid] = acc[e]; __syncthreads();
    for (int s = 128; s > 0; s >>= 1) { if (tid < s) red[tid] += red[tid + s]; __syncthreads(); }
    if (tid == 0) logits[e] = red[0];
    __syncthreads();
  }
  if (tid == 0) {
    int best = 0; float bv = logits[0];
    for (int e = 1; e < E_; ++e) if (logits[e] > bv) { bv = logits[e]; best = e; }
    top_idx[t]  = best;
    top_gate[t] = 1.f / (1.f + __expf(-bv));
  }
}

__global__ void zero_counts(int* c) { if (threadIdx.x < E_) c[threadIdx.x] = 0; }

__global__ __launch_bounds__(256)
void build_lists(const int* __restrict__ idx, int* __restrict__ counts,
                 int* __restrict__ toklist) {
  int t = blockIdx.x * 256 + threadIdx.x;
  if (t < T_) {
    int e = idx[t];
    int s = atomicAdd(&counts[e], 1);
    toklist[e * T_ + s] = t;     // slot order nondeterministic; per-token values deterministic
  }
}

__global__ void calc_offsets(const int* __restrict__ counts, int* __restrict__ offsets) {
  if (threadIdx.x == 0) {
    int s = 0;
    for (int e = 0; e < E_; ++e) { offsets[e] = s; s += counts[e]; }
  }
}

__global__ __launch_bounds__(256)
void silu_mul(const float* __restrict__ g, const float* __restrict__ u,
              __bf16* __restrict__ h, int n) {
  int i = blockIdx.x * 256 + threadIdx.x;
  if (i < n) {
    float gv = g[i];
    float s = gv / (1.f + __expf(-gv));
    h[i] = (__bf16)(s * u[i]);
  }
}

// ---------------------------------------------------------------------------
// Launch
// ---------------------------------------------------------------------------
extern "C" void kernel_launch(void* const* d_in, const int* in_sizes, int n_in,
                              void* d_out, int out_size, void* d_ws, size_t ws_size,
                              hipStream_t stream) {
  (void)in_sizes; (void)n_in; (void)out_size; (void)ws_size;
  const int*   positions = (const int*)d_in[0];
  const float* hidden    = (const float*)d_in[1];
  const float* ln1_w     = (const float*)d_in[2];
  const float* ln2_w     = (const float*)d_in[3];
  const float* w_qkv     = (const float*)d_in[4];
  const float* w_o       = (const float*)d_in[5];
  const float* q_norm_w  = (const float*)d_in[6];
  const float* k_norm_w  = (const float*)d_in[7];
  const float* router_w  = (const float*)d_in[8];
  const float* ws_gate   = (const float*)d_in[9];
  const float* ws_up     = (const float*)d_in[10];
  const float* ws_down   = (const float*)d_in[11];
  const float* we_gate   = (const float*)d_in[12];
  const float* we_up     = (const float*)d_in[13];
  const float* we_down   = (const float*)d_in[14];
  float* out = (float*)d_out;

  uintptr_t p = (uintptr_t)d_ws;
  auto carve = [&](size_t bytes) -> void* {
    void* r = (void*)p;
    p += (bytes + 255) & ~(size_t)255;
    return r;
  };
  float*  qkv     = (float*) carve((size_t)T_ * QKV_ * 4);
  __bf16* xn1bf   = (__bf16*)carve((size_t)T_ * H_ * 2);
  __bf16* qbf     = (__bf16*)carve((size_t)T_ * QS_ * 2);
  __bf16* kbf     = (__bf16*)carve((size_t)T_ * KVS_ * 2);
  __bf16* vbf     = (__bf16*)carve((size_t)T_ * KVS_ * 2);
  __bf16* attnbf  = (__bf16*)carve((size_t)T_ * QS_ * 2);
  float*  x2      = (float*) carve((size_t)T_ * H_ * 4);
  __bf16* xn2bf   = (__bf16*)carve((size_t)T_ * H_ * 2);
  float*  gbuf    = (float*) carve((size_t)T_ * I_ * 4);
  float*  ubuf    = (float*) carve((size_t)T_ * I_ * 4);
  __bf16* hbuf    = (__bf16*)carve((size_t)T_ * I_ * 2);
  int*    top_idx = (int*)   carve((size_t)T_ * 4);
  float*  top_gate= (float*) carve((size_t)T_ * 4);
  int*    counts  = (int*)   carve(E_ * 4);
  int*    offsets = (int*)   carve(E_ * 4);
  int*    toklist = (int*)   carve((size_t)E_ * T_ * 4);

  // 1) ln1 -> bf16
  rmsnorm_bf16<<<T_, 256, 0, stream>>>(hidden, H_, 0, ln1_w, xn1bf, H_, H_);
  // 2) qkv = xn1 @ w_qkv
  gemm_wmma<<<dim3(QKV_ / 128, T_ / 128), 256, 0, stream>>>(
      xn1bf, w_qkv, qkv, T_, QKV_, H_, H_, QKV_, QKV_,
      nullptr, nullptr, nullptr, nullptr, nullptr, nullptr, 0);
  // 3) rope in-place on q,k
  rope_kernel<<<dim3(T_, NH_ + NKV_), 64, 0, stream>>>(positions, qkv);
  // 4) q/k rmsnorm -> bf16, v -> bf16
  rmsnorm_bf16<<<T_, 256, 0, stream>>>(qkv, QKV_, 0,   q_norm_w, qbf, QS_,  QS_);
  rmsnorm_bf16<<<T_, 256, 0, stream>>>(qkv, QKV_, QS_, k_norm_w, kbf, KVS_, KVS_);
  conv_v<<<(T_ * KVS_ + 255) / 256, 256, 0, stream>>>(qkv, vbf);
  // 5) attention
  flash_attn<<<dim3(T_ / 128, NH_), 256, 0, stream>>>(qbf, kbf, vbf, attnbf);
  // 6) x2 = attn @ w_o + hidden
  gemm_wmma<<<dim3(H_ / 128, T_ / 128), 256, 0, stream>>>(
      attnbf, w_o, x2, T_, H_, QS_, QS_, H_, H_,
      nullptr, nullptr, nullptr, nullptr, nullptr, hidden, 0);
  // 7) ln2 -> bf16
  rmsnorm_bf16<<<T_, 256, 0, stream>>>(x2, H_, 0, ln2_w, xn2bf, H_, H_);
  // 8) router top-1
  router_kernel<<<T_, 256, 0, stream>>>(xn2bf, router_w, top_idx, top_gate);
  zero_counts<<<1, 32, 0, stream>>>(counts);
  build_lists<<<(T_ + 255) / 256, 256, 0, stream>>>(top_idx, counts, toklist);
  calc_offsets<<<1, 32, 0, stream>>>(counts, offsets);
  // 9) shared expert
  gemm_wmma<<<dim3(I_ / 128, T_ / 128), 256, 0, stream>>>(
      xn2bf, ws_gate, gbuf, T_, I_, H_, H_, I_, I_,
      nullptr, nullptr, nullptr, nullptr, nullptr, nullptr, 0);
  gemm_wmma<<<dim3(I_ / 128, T_ / 128), 256, 0, stream>>>(
      xn2bf, ws_up, ubuf, T_, I_, H_, H_, I_, I_,
      nullptr, nullptr, nullptr, nullptr, nullptr, nullptr, 0);
  silu_mul<<<((size_t)T_ * I_ + 255) / 256, 256, 0, stream>>>(gbuf, ubuf, hbuf, T_ * I_);
  // out = x2 + shared_down
  gemm_wmma<<<dim3(H_ / 128, T_ / 128), 256, 0, stream>>>(
      hbuf, ws_down, out, T_, H_, I_, I_, H_, H_,
      nullptr, nullptr, nullptr, nullptr, nullptr, x2, 0);
  // 10) routed experts: gathered gate/up per expert into compacted rows
  for (int e = 0; e < E_; ++e) {
    gemm_wmma<<<dim3(I_ / 128, T_ / 128), 256, 0, stream>>>(
        xn2bf, we_gate + (size_t)e * H_ * I_, gbuf, T_, I_, H_, H_, I_, I_,
        toklist + e * T_, offsets + e, counts + e, nullptr, nullptr, nullptr, 0);
    gemm_wmma<<<dim3(I_ / 128, T_ / 128), 256, 0, stream>>>(
        xn2bf, we_up + (size_t)e * H_ * I_, ubuf, T_, I_, H_, H_, I_, I_,
        toklist + e * T_, offsets + e, counts + e, nullptr, nullptr, nullptr, 0);
  }
  silu_mul<<<((size_t)T_ * I_ + 255) / 256, 256, 0, stream>>>(gbuf, ubuf, hbuf, T_ * I_);
  // 11) expert down, scatter rows back to tokens with sigmoid gate, out += ...
  for (int e = 0; e < E_; ++e) {
    gemm_wmma<<<dim3(H_ / 128, T_ / 128), 256, 0, stream>>>(
        hbuf, we_down + (size_t)e * I_ * H_, out, T_, H_, I_, I_, H_, H_,
        nullptr, offsets + e, counts + e, toklist + e * T_, top_gate, nullptr, 1);
  }
}
